// LlamaMLP_62801011802988
// MI455X (gfx1250) — matrix operand order repaired and checked by reading, not compiled, not run on hardware
//
#include <hip/hip_runtime.h>
#include <hip/hip_bf16.h>

typedef __attribute__((ext_vector_type(16))) _Float16 v16h;
typedef __attribute__((ext_vector_type(8)))  _Float16 v8h;
typedef __attribute__((ext_vector_type(8)))  float    v8f;

union Frag { v16h v; v8h h[2]; };

#define BL_EPS 1e-5f

// async global->LDS b128 (GVS mode: SGPR64 base + per-lane u32 byte offset).
// INST_OFFSET applies to both the LDS and global addresses (ISA 08 §4.4), so
// consecutive 16B chunks reuse one lds/voff register pair.
#define ASYNC_LD_B128(ldsaddr, voff, sbase, off)                               \
    asm volatile("global_load_async_to_lds_b128 %0, %1, %2 offset:%3"          \
                 :: "v"(ldsaddr), "v"(voff), "s"(sbase), "i"(off) : "memory")

#define WAIT_ASYNC_0() asm volatile("s_wait_asynccnt 0x0" ::: "memory")

// ---------------- scale = mean(|w|): two-pass deterministic reduction ----------------
__global__ void absum_partial(const float* __restrict__ w, size_t n,
                              float* __restrict__ partial) {
    __shared__ float sm[256];
    float s = 0.f;
    const size_t stride = (size_t)gridDim.x * blockDim.x;
    for (size_t i = (size_t)blockIdx.x * blockDim.x + threadIdx.x; i < n; i += stride)
        s += fabsf(w[i]);
    sm[threadIdx.x] = s;
    __syncthreads();
    for (int o = 128; o > 0; o >>= 1) {
        if ((int)threadIdx.x < o) sm[threadIdx.x] += sm[threadIdx.x + o];
        __syncthreads();
    }
    if (threadIdx.x == 0) partial[blockIdx.x] = sm[0];
}

__global__ void absum_final(const float* __restrict__ partial, int np,
                            float inv_n, float* __restrict__ out) {
    __shared__ float sm[256];
    float s = 0.f;
    for (int i = threadIdx.x; i < np; i += 256) s += partial[i];
    sm[threadIdx.x] = s;
    __syncthreads();
    for (int o = 128; o > 0; o >>= 1) {
        if ((int)threadIdx.x < o) sm[threadIdx.x] += sm[threadIdx.x + o];
        __syncthreads();
    }
    if (threadIdx.x == 0) out[0] = sm[0] * inv_n;
}

// ---------------- ternarize weights to f16 {-1,0,+1} ----------------
__global__ void quantize_w(const float* __restrict__ w,
                           const float* __restrict__ scale,
                           _Float16* __restrict__ q, size_t n) {
    const float inv = 1.0f / (scale[0] + BL_EPS);
    const size_t stride = (size_t)gridDim.x * blockDim.x;
    for (size_t i = (size_t)blockIdx.x * blockDim.x + threadIdx.x; i < n; i += stride) {
        float v = w[i] * inv;
        v = fminf(1.0f, fmaxf(-1.0f, v));
        q[i] = (_Float16)rintf(v);   // round-half-even, matches jnp.round
    }
}

__global__ void cvt_f16(const float* __restrict__ x, _Float16* __restrict__ y, size_t n) {
    const size_t stride = (size_t)gridDim.x * blockDim.x;
    for (size_t i = (size_t)blockIdx.x * blockDim.x + threadIdx.x; i < n; i += stride)
        y[i] = (_Float16)x[i];
}

// ---------------- tiled WMMA GEMM: C = s * (A[M,K] @ B[N,K].T) ----------------
// FUSED: C_f16 = (s0*s1) * (A@B0.T) * (A@B1.T)   (gate*up, shared A tile)
// else : C_f32 = s0 * (A@B0.T)                   (down proj)
// Block: 256 threads (8 waves), tile 64(M) x 128(N), K-step 64, double-buffered
// LDS fed by GLOBAL_LOAD_ASYNC_TO_LDS_B128 (ASYNCcnt pipeline).
// Wave (wm 0..1, wn 0..3) owns a 32x32 register tile = 2x2 WMMA frags.
template<bool FUSED>
__global__ void __launch_bounds__(256)
bitlinear_gemm(const _Float16* __restrict__ A,
               const _Float16* __restrict__ B0,
               const _Float16* __restrict__ B1,
               void* __restrict__ Cout,
               const float* __restrict__ sc,
               int M, int N, int K)
{
    constexpr int KS  = 64;            // K chunk
    constexpr int ASZ = 64 * KS;       // A tile elems per buffer
    constexpr int BSZ = 128 * KS;      // B tile elems per buffer

    __shared__ _Float16 As [2 * ASZ];
    __shared__ _Float16 B0s[2 * BSZ];
    __shared__ _Float16 B1s[FUSED ? 2 * BSZ : 32];

    const int tid  = threadIdx.x;
    const int lane = tid & 31;
    const int wave = tid >> 5;
    const int wm   = wave >> 2;        // 0..1 : 32-row strip
    const int wn   = wave & 3;         // 0..3 : 32-col strip
    const int l16  = lane & 15;
    const int hi   = lane >> 4;        // lane half, per ISA fragment layout

    const int blockM = blockIdx.x * 64;
    const int blockN = blockIdx.y * 128;

    v8f acc0[2][2] = {};
    v8f acc1[2][2] = {};               // dead when !FUSED

    // ---- async staging geometry ----
    // A tile 64x64: thread -> row tid>>2, col (tid&3)*16 ; 2 x b128
    // B tile 128x64: thread -> row tid>>1, col (tid&1)*32 ; 4 x b128
    const int arr = tid >> 2, acc_ = (tid & 3) << 4;
    const int brr = tid >> 1, bcc  = (tid & 1) << 5;
    const uint32_t aVoff0 = (uint32_t)(((size_t)(blockM + arr) * K + acc_) * 2);
    const uint32_t bVoff0 = (uint32_t)(((size_t)(blockN + brr) * K + bcc) * 2);
    const uint32_t aLds0  = (uint32_t)(uintptr_t)&As [(size_t)arr * KS + acc_];
    const uint32_t bLds0  = (uint32_t)(uintptr_t)&B0s[(size_t)brr * KS + bcc];
    const uint32_t cLds0  = FUSED ? (uint32_t)(uintptr_t)&B1s[(size_t)brr * KS + bcc] : 0u;

    auto issue = [&](int kk, int buf) {
        const uint32_t av = aVoff0 + (uint32_t)kk * 2;
        const uint32_t bv = bVoff0 + (uint32_t)kk * 2;
        const uint32_t al = aLds0 + (uint32_t)(buf ? ASZ * 2 : 0);
        const uint32_t bl = bLds0 + (uint32_t)(buf ? BSZ * 2 : 0);
        ASYNC_LD_B128(al, av, A, 0);
        ASYNC_LD_B128(al, av, A, 16);
        ASYNC_LD_B128(bl, bv, B0, 0);
        ASYNC_LD_B128(bl, bv, B0, 16);
        ASYNC_LD_B128(bl, bv, B0, 32);
        ASYNC_LD_B128(bl, bv, B0, 48);
        if constexpr (FUSED) {
            const uint32_t cl = cLds0 + (uint32_t)(buf ? BSZ * 2 : 0);
            ASYNC_LD_B128(cl, bv, B1, 0);
            ASYNC_LD_B128(cl, bv, B1, 16);
            ASYNC_LD_B128(cl, bv, B1, 32);
            ASYNC_LD_B128(cl, bv, B1, 48);
        }
    };

    const int NC = K / KS;
    issue(0, 0);

    for (int c = 0; c < NC; ++c) {
        const int buf = c & 1;
        WAIT_ASYNC_0();            // my chunk-c writes landed in LDS
        __syncthreads();           // everyone's landed; prev reads of buf^1 done
        if (c + 1 < NC) issue((c + 1) * KS, buf ^ 1);

        const _Float16* Asb = As  + (size_t)buf * ASZ;
        const _Float16* B0b = B0s + (size_t)buf * BSZ;
        const _Float16* B1b = FUSED ? (B1s + (size_t)buf * BSZ) : nullptr;

#pragma unroll
        for (int ks = 0; ks < KS; ks += 32) {
            Frag a[2], b0[2], b1[2];
#pragma unroll
            for (int i = 0; i < 2; ++i) {
                // A 16x32 f16: lane holds row M=l16, K = hi*16 + 0..15 contiguous
                const int r = (wm * 32 + i * 16 + l16) * KS + ks;
                a[i].h[0] = *(const v8h*)&Asb[r + hi * 16];
                a[i].h[1] = *(const v8h*)&Asb[r + hi * 16 + 8];
            }
#pragma unroll
            for (int j = 0; j < 2; ++j) {
                // B 32x16 f16: lane holds col N=l16, K = hi*16 + 0..15 contiguous
                const int r = (wn * 32 + j * 16 + l16) * KS + ks;
                b0[j].h[0] = *(const v8h*)&B0b[r + hi * 16];
                b0[j].h[1] = *(const v8h*)&B0b[r + hi * 16 + 8];
                if constexpr (FUSED) {
                    b1[j].h[0] = *(const v8h*)&B1b[r + hi * 16];
                    b1[j].h[1] = *(const v8h*)&B1b[r + hi * 16 + 8];
                }
            }
#pragma unroll
            for (int i = 0; i < 2; ++i)
#pragma unroll
                for (int j = 0; j < 2; ++j) {
                    acc0[i][j] = __builtin_amdgcn_wmma_f32_16x16x32_f16(
                        false, a[i].v, false, b0[j].v, (short)0, acc0[i][j], false, false);
                    if constexpr (FUSED)
                        acc1[i][j] = __builtin_amdgcn_wmma_f32_16x16x32_f16(
                            false, a[i].v, false, b1[j].v, (short)0, acc1[i][j], false, false);
                }
        }
    }

    // epilogue: C frag layout = 8 VGPRs, m = v + hi*8, n = l16
    const float s = FUSED ? sc[0] * sc[1] : sc[0];
#pragma unroll
    for (int i = 0; i < 2; ++i)
#pragma unroll
        for (int j = 0; j < 2; ++j) {
            const int mb = blockM + wm * 32 + i * 16 + hi * 8;
            const int nb = blockN + wn * 32 + j * 16 + l16;
#pragma unroll
            for (int v = 0; v < 8; ++v) {
                const size_t idx = (size_t)(mb + v) * (size_t)N + nb;
                if constexpr (FUSED)
                    ((_Float16*)Cout)[idx] = (_Float16)(s * acc0[i][j][v] * acc1[i][j][v]);
                else
                    ((float*)Cout)[idx] = s * acc0[i][j][v];
            }
        }
}

extern "C" void kernel_launch(void* const* d_in, const int* in_sizes, int n_in,
                              void* d_out, int out_size, void* d_ws, size_t ws_size,
                              hipStream_t stream) {
    (void)n_in; (void)out_size; (void)ws_size;
    const float* x  = (const float*)d_in[0];   // [M,H] fp32, M = B*S
    const float* wg = (const float*)d_in[1];   // [I,H]
    const float* wu = (const float*)d_in[2];   // [I,H]
    const float* wd = (const float*)d_in[3];   // [H,I]

    const int H = 4096, I = 11008;
    const int M = in_sizes[0] / H;                 // 8192
    const size_t nW     = (size_t)I * H;
    const size_t nX     = (size_t)M * H;
    const size_t nInter = (size_t)M * I;

    char* ws = (char*)d_ws;
    float* scales   = (float*)ws;                  // [0]=gate [1]=up [2]=down
    float* partials = (float*)(ws + 256);          // 1024 floats
    size_t off = 256 + 4096;
    auto al = [](size_t v) { return (v + 255) & ~(size_t)255; };
    _Float16* xh    = (_Float16*)(ws + off); off = al(off + nX * 2);
    _Float16* qg    = (_Float16*)(ws + off); off = al(off + nW * 2);
    _Float16* qu    = (_Float16*)(ws + off); off = al(off + nW * 2);
    _Float16* qd    = (_Float16*)(ws + off); off = al(off + nW * 2);
    _Float16* inter = (_Float16*)(ws + off); off = al(off + nInter * 2);

    const float* wsrc[3] = {wg, wu, wd};
    _Float16*    qdst[3] = {qg, qu, qd};
    for (int i = 0; i < 3; ++i) {
        absum_partial<<<1024, 256, 0, stream>>>(wsrc[i], nW, partials);
        absum_final<<<1, 256, 0, stream>>>(partials, 1024, 1.0f / (float)nW, scales + i);
        quantize_w<<<4096, 256, 0, stream>>>(wsrc[i], scales + i, qdst[i], nW);
    }
    cvt_f16<<<4096, 256, 0, stream>>>(x, xh, nX);

    // inter = (sg*su) * (x@qg.T) * (x@qu.T)    [M,I] f16
    dim3 g1(M / 64, I / 128);
    bitlinear_gemm<true><<<g1, 256, 0, stream>>>(xh, qg, qu, (void*)inter, scales, M, I, H);

    // out = sd * (inter@qd.T)                  [M,H] f32
    dim3 g2(M / 64, H / 128);
    bitlinear_gemm<false><<<g2, 256, 0, stream>>>(inter, qd, nullptr, d_out, scales + 2, M, H, I);
}
